// SelfAttention_13881334301124
// MI455X (gfx1250) — matrix operand order repaired and checked by reading, not compiled, not run on hardware
//
#include <hip/hip_runtime.h>

// ---------------------------------------------------------------------------
// MI455X / gfx1250 fused relative-position self-attention, round 5:
// - f16 WMMA (v_wmma_f32_16x16x32_f16), f32 accum
// - attention: 4 waves / WG share K(64x32) + V(32x64) LDS tiles per 32-key
//   block (4x less L2 traffic); tiles filled with GLOBAL_LOAD_ASYNC_TO_LDS
//   (ASYNCcnt-tracked) when the builtin exists, sync b128 copies otherwise
// - K stored transposed; E^T (padded) and Wo^T pre-materialized in f16
// - all global result stores staged through LDS -> coalesced b128 stores
// ---------------------------------------------------------------------------

typedef __attribute__((ext_vector_type(16))) _Float16 v16h;
typedef __attribute__((ext_vector_type(8)))  _Float16 v8h;
typedef __attribute__((ext_vector_type(8)))  float    v8f;
typedef __attribute__((__vector_size__(16))) int      avint4;  // raw vector, not HIP int4

constexpr int N_ = 4, S_ = 2048, EMB_ = 512, H_ = 8, D_ = 64;
constexpr int SP_ = S_ + 64;  // padded E^T row length

#if __has_builtin(__builtin_amdgcn_global_load_async_to_lds_b128) && \
    __has_builtin(__builtin_amdgcn_s_wait_asynccnt)
#define USE_ASYNC 1
#else
#define USE_ASYNC 0
#endif

// copy 8 halves (16B) global -> LDS
__device__ __forceinline__ void cp8(_Float16* l, const _Float16* g) {
#if USE_ASYNC
  __builtin_amdgcn_global_load_async_to_lds_b128((avint4*)g, (avint4*)l, 0, 0);
#else
  *(int4*)l = *(const int4*)g;
#endif
}
__device__ __forceinline__ void cp_fence() {
#if USE_ASYNC
  __builtin_amdgcn_s_wait_asynccnt(0);
#endif
}

__device__ __forceinline__ v8f wmma_f16(v16h a, v16h b, v8f c) {
  return __builtin_amdgcn_wmma_f32_16x16x32_f16(
      false, a, false, b, (short)0, c, false, false);
}

// --- fragment loaders (layouts per CDNA5 ISA 7.12.2, wave32) ----------------
// A 16x32 (MxK) 16-bit: row M = lane&15; K = (hi&7) + 8*(lane>>4) + 16*(hi>>3)
__device__ __forceinline__ v16h a_frag_f16(const _Float16* src, int ld, int lane) {
  const _Float16* p = src + (lane & 15) * ld + ((lane >> 4) << 3);
  union { v16h v; int4 q[2]; } u;
  u.q[0] = *(const int4*)p;
  u.q[1] = *(const int4*)(p + 16);
  return u.v;
}
__device__ __forceinline__ v16h a_frag_f32(const float* src, int ld, int lane) {
  const float* p = src + (lane & 15) * ld + ((lane >> 4) << 3);
  float4 f0 = *(const float4*)p;
  float4 f1 = *(const float4*)(p + 4);
  float4 f2 = *(const float4*)(p + 16);
  float4 f3 = *(const float4*)(p + 20);
  v16h a;
  a[0]  = (_Float16)f0.x; a[1]  = (_Float16)f0.y; a[2]  = (_Float16)f0.z; a[3]  = (_Float16)f0.w;
  a[4]  = (_Float16)f1.x; a[5]  = (_Float16)f1.y; a[6]  = (_Float16)f1.z; a[7]  = (_Float16)f1.w;
  a[8]  = (_Float16)f2.x; a[9]  = (_Float16)f2.y; a[10] = (_Float16)f2.z; a[11] = (_Float16)f2.w;
  a[12] = (_Float16)f3.x; a[13] = (_Float16)f3.y; a[14] = (_Float16)f3.z; a[15] = (_Float16)f3.w;
  return a;
}
// B 32x16 (KxN): N = lane&15; K = (hi&7) + 8*(lane>>4) + 16*(hi>>3).
__device__ __forceinline__ v16h b_frag_pt(const _Float16* p) {
  union { v16h v; int4 q[2]; } u;
  u.q[0] = *(const int4*)p;
  u.q[1] = *(const int4*)(p + 8);
  return u.v;
}
__device__ __forceinline__ v16h b_frag_f16_vec(const _Float16* src, int ld, int lane) {
  v16h b;
#pragma unroll
  for (int i = 0; i < 16; ++i)
    b[i] = src[(size_t)((i & 7) + ((lane >> 4) << 3) + ((i >> 3) << 4)) * ld + (lane & 15)];
  return b;
}
// unaligned base (E^T band window) -> scalar halves
__device__ __forceinline__ v16h b_frag_f16_s(const _Float16* src, int ld, int lane) {
  const _Float16* p = src + (lane & 15);
  v16h b;
#pragma unroll
  for (int i = 0; i < 16; ++i) b[i] = p[(size_t)((i & 7) + ((lane >> 4) << 3) + ((i >> 3) << 4)) * ld];
  return b;
}
// transposed pattern from f32 (tiny 64x64 QKV weights only)
__device__ __forceinline__ v16h b_frag_T_f32(const float* src, int ld, int lane) {
  v16h b;
  int nn = lane & 15;
#pragma unroll
  for (int i = 0; i < 16; ++i) b[i] = (_Float16)src[nn * ld + (i & 7) + ((lane >> 4) << 3) + ((i >> 3) << 4)];
  return b;
}

// ---------------------------------------------------------------------------
// Prep kernels: E^T padded ([D][S+64], top-clamped) and Wo^T, both f16.
// ---------------------------------------------------------------------------
__global__ void prep_E_kernel(const float* __restrict__ E, _Float16* __restrict__ EtPad) {
  int i = blockIdx.x * blockDim.x + threadIdx.x;
  if (i >= D_ * SP_) return;
  int d = i / SP_, e = i % SP_;
  int es = e < S_ ? e : S_ - 1;
  EtPad[i] = (_Float16)E[es * D_ + d];
}
__global__ void prep_Wo_kernel(const float* __restrict__ Wo, _Float16* __restrict__ WoT) {
  int i = blockIdx.x * blockDim.x + threadIdx.x;
  if (i >= EMB_ * EMB_) return;
  int k = i / EMB_, n = i % EMB_;
  WoT[i] = (_Float16)Wo[n * EMB_ + k];
}

// ---------------------------------------------------------------------------
// Kernel 1: Q/K/V projection. One wave per (n, h, 16 s-rows).
// ---------------------------------------------------------------------------
__global__ __launch_bounds__(32) void proj_kernel(
    const float* __restrict__ x,
    const float* __restrict__ Wq, const float* __restrict__ bq,
    const float* __restrict__ Wk, const float* __restrict__ bk,
    const float* __restrict__ Wv, const float* __restrict__ bv,
    _Float16* __restrict__ q16, _Float16* __restrict__ kT,
    _Float16* __restrict__ v16p) {
  __shared__ _Float16 stg[16 * 64];
  int lane = threadIdx.x;
  int b = blockIdx.x;
  int st = b & 127, h = (b >> 7) & 7, n = b >> 10;
  int s0 = st * 16;
  int col = lane & 15, rhi = (lane >> 4) << 3;

  const float* xb = x + ((size_t)n * S_ + s0) * EMB_ + h * D_;
  v16h a0 = a_frag_f32(xb, EMB_, lane);
  v16h a1 = a_frag_f32(xb + 32, EMB_, lane);

  size_t obase = ((size_t)(n * H_ + h) * S_ + s0) * D_;
  size_t ktbase = (size_t)(n * H_ + h) * D_ * S_;
  const float* Ws[3] = {Wq, Wk, Wv};
  const float* bs[3] = {bq, bk, bv};

#pragma unroll
  for (int w = 0; w < 3; ++w) {
    v8f acc[4];
#pragma unroll
    for (int j = 0; j < 4; ++j) {
      float bias = bs[w][16 * j + col];
#pragma unroll
      for (int r = 0; r < 8; ++r) acc[j][r] = bias;
      v16h b0 = b_frag_T_f32(Ws[w] + 16 * j * D_, D_, lane);
      v16h b1 = b_frag_T_f32(Ws[w] + 16 * j * D_ + 32, D_, lane);
      acc[j] = wmma_f16(a0, b0, acc[j]);
      acc[j] = wmma_f16(a1, b1, acc[j]);
    }
    if (w == 1) {
      // K transposed: lane's 8 rows = consecutive s for fixed d -> 16B store
#pragma unroll
      for (int j = 0; j < 4; ++j) {
        v8h kv;
#pragma unroll
        for (int r = 0; r < 8; ++r) kv[r] = (_Float16)acc[j][r];
        *(v8h*)(kT + ktbase + (size_t)(16 * j + col) * S_ + s0 + rhi) = kv;
      }
    } else {
#pragma unroll
      for (int j = 0; j < 4; ++j)
#pragma unroll
        for (int r = 0; r < 8; ++r)
          stg[(r + rhi) * 64 + 16 * j + col] = (_Float16)acc[j][r];
      _Float16* outp = (w == 0 ? q16 : v16p) + obase;
      const int4* s4 = (const int4*)stg;
      int row = lane >> 1, co = (lane & 1) * 32;
      _Float16* gp = outp + (size_t)row * D_ + co;
#pragma unroll
      for (int t = 0; t < 4; ++t)
        *(int4*)(gp + 8 * t) = s4[lane * 4 + t];
    }
  }
}

// ---------------------------------------------------------------------------
// Kernel 2: fused attention. 4 waves/WG, 64 q-rows; K/V tiles shared via LDS.
// ---------------------------------------------------------------------------
__global__ __launch_bounds__(128) void attn_kernel(
    const _Float16* __restrict__ q16, const _Float16* __restrict__ kT,
    const _Float16* __restrict__ v16p, const _Float16* __restrict__ EtPad,
    _Float16* __restrict__ z16) {
  __shared__ _Float16 ktile[64 * 32];     // [kd][key]   4KB
  __shared__ _Float16 vtile[32 * 64];     // [key][d]    4KB
  __shared__ float    qe_lds[4][16 * 48];
  __shared__ _Float16 p_lds[4][16 * 32];
  __shared__ _Float16 zstg[4][16 * 64];

  int tid = threadIdx.x;
  int lane = tid & 31, wid = tid >> 5;
  int b = blockIdx.x;
  int qb = b & 31, h = (b >> 5) & 7, n = b >> 8;   // S/64 = 32 q-blocks
  int q0 = qb * 64 + wid * 16;
  int col = lane & 15, rhi = (lane >> 4) << 3;
  int klx = (lane & 15) + ((lane >> 4) << 4);      // B-fragment K index

  const _Float16* qbp = q16  + ((size_t)(n * H_ + h) * S_ + q0) * D_;
  const _Float16* kTb = kT   + (size_t)(n * H_ + h) * D_ * S_;
  const _Float16* vb  = v16p + (size_t)(n * H_ + h) * S_ * D_;

  v16h aq0 = a_frag_f16(qbp, D_, lane);
  v16h aq1 = a_frag_f16(qbp + 32, D_, lane);

  float m[8], l[8];
  v8f accP[4] = {}, accS[4] = {};
#pragma unroll
  for (int r = 0; r < 8; ++r) { m[r] = -3.0e38f; l[r] = 0.f; }

  const float scale = 0.044194173824159216f;  // 1/sqrt(EMB)
  float*    qel = qe_lds[wid];
  _Float16* pl  = p_lds[wid];

  // cooperative fill mapping (128 threads)
  int krow = tid >> 1, kco = (tid & 1) * 16;   // K tile: 64 rows x 32 halves
  int vrow = tid >> 2, vco = (tid & 3) * 16;   // V tile: 32 rows x 64 halves

  for (int kbk = 0; kbk < S_ / 32; ++kbk) {
    int c0 = kbk * 32;
    __syncthreads();  // previous tile fully consumed
    cp8(ktile + krow * 32 + kco,     kTb + (size_t)krow * S_ + c0 + kco);
    cp8(ktile + krow * 32 + kco + 8, kTb + (size_t)krow * S_ + c0 + kco + 8);
    cp8(vtile + vrow * 64 + vco,     vb + (size_t)(c0 + vrow) * D_ + vco);
    cp8(vtile + vrow * 64 + vco + 8, vb + (size_t)(c0 + vrow) * D_ + vco + 8);
    cp_fence();
    __syncthreads();  // tiles visible to all waves

    if (c0 + 32 < S_)
      __builtin_prefetch((const void*)(vb + (size_t)(c0 + 32 + vrow) * D_ + vco), 0, 1);

    bool doP = (c0 + 31 > q0);      // wave-uniform
    bool doS = (c0 <= q0 + 15);

    if (doP) {
      v8f s0 = {}, s1 = {};
      v16h bkA = b_frag_f16_s(ktile, 32, lane);             // kd 0..31, keys 0..15
      v16h bkB = b_frag_f16_s(ktile + 32 * 32, 32, lane);      // kd 32..63
      v16h bkC = b_frag_f16_s(ktile + 16, 32, lane);        // keys 16..31
      v16h bkD = b_frag_f16_s(ktile + 32 * 32 + 16, 32, lane);
      s0 = wmma_f16(aq0, bkA, s0); s0 = wmma_f16(aq1, bkB, s0);
      s1 = wmma_f16(aq0, bkC, s1); s1 = wmma_f16(aq1, bkD, s1);

#pragma unroll
      for (int r = 0; r < 8; ++r) {
        int qrow = q0 + r + rhi;
        float sv0 = (c0 + col > qrow)      ? s0[r] * scale : -1e30f;
        float sv1 = (c0 + 16 + col > qrow) ? s1[r] * scale : -1e30f;
        float tmax = fmaxf(sv0, sv1);
#pragma unroll
        for (int off = 8; off >= 1; off >>= 1)
          tmax = fmaxf(tmax, __shfl_xor(tmax, off, 32));
        float nm = fmaxf(m[r], tmax);
        float corr = __expf(m[r] - nm);
        float p0 = __expf(sv0 - nm);
        float p1 = __expf(sv1 - nm);
        float ts = p0 + p1;
#pragma unroll
        for (int off = 8; off >= 1; off >>= 1)
          ts += __shfl_xor(ts, off, 32);
        m[r] = nm;
        l[r] = l[r] * corr + ts;
#pragma unroll
        for (int j = 0; j < 4; ++j) accP[j][r] *= corr;
        int row = r + rhi;
        pl[row * 32 + col]      = (_Float16)p0;
        pl[row * 32 + 16 + col] = (_Float16)p1;
      }

      v16h ap;  // C-layout -> A-layout via per-wave LDS
      {
        const _Float16* pp = pl + (lane & 15) * 32 + rhi;
        union { v16h v; int4 q[2]; } u;
        u.q[0] = *(const int4*)pp;
        u.q[1] = *(const int4*)(pp + 16);
        ap = u.v;
      }
#pragma unroll
      for (int j = 0; j < 4; ++j) {
        v16h bv = b_frag_f16_s(vtile + 16 * j, 64, lane);
        accP[j] = wmma_f16(ap, bv, accP[j]);
      }
    }

    if (doS) {
      // banded QE over E rows [e0, e0+48); EtPad top-clamp makes it safe
      int e0 = (S_ - 1) - q0 + c0 - 15;
      v8f qe[3] = {};
#pragma unroll
      for (int t = 0; t < 3; ++t) {
        v16h be0 = b_frag_f16_s(EtPad + e0 + 16 * t, SP_, lane);
        v16h be1 = b_frag_f16_s(EtPad + (size_t)32 * SP_ + e0 + 16 * t, SP_, lane);
        qe[t] = wmma_f16(aq0, be0, qe[t]);
        qe[t] = wmma_f16(aq1, be1, qe[t]);
      }
#pragma unroll
      for (int t = 0; t < 3; ++t)
#pragma unroll
        for (int r = 0; r < 8; ++r)
          qel[(r + rhi) * 48 + 16 * t + col] = qe[t][r];

      // diagonal gather: Smat(row, key) = QE[row][key - row + 15] for c <= q
      v16h as;
      {
        int row = lane & 15;
        int qrow = q0 + row;
#pragma unroll
        for (int i = 0; i < 16; ++i) {
          int K = (i & 7) + ((i >> 3) << 4) + rhi;
          float v = (c0 + K <= qrow) ? qel[row * 48 + (K - row + 15)] : 0.f;
          as[i] = (_Float16)v;
        }
      }
#pragma unroll
      for (int j = 0; j < 4; ++j) {
        v16h bv = b_frag_f16_s(vtile + 16 * j, 64, lane);
        accS[j] = wmma_f16(as, bv, accS[j]);
      }
    }
  }

  // z = accP/l + accS -> per-wave LDS stage -> coalesced b128 stores
  _Float16* zs = zstg[wid];
#pragma unroll
  for (int j = 0; j < 4; ++j)
#pragma unroll
    for (int r = 0; r < 8; ++r)
      zs[(r + rhi) * 64 + 16 * j + col] =
          (_Float16)(accP[j][r] / l[r] + accS[j][r]);
  {
    const int4* z4 = (const int4*)zs;
    int row = lane >> 1, co = (lane & 1) * 32;
    _Float16* zp = z16 + ((size_t)(n * S_ + q0 + row)) * EMB_ + h * D_ + co;
#pragma unroll
    for (int t = 0; t < 4; ++t)
      *(int4*)(zp + 8 * t) = z4[lane * 4 + t];
  }
}

// ---------------------------------------------------------------------------
// Kernel 3: out = z @ Wo.T + bo. One wave per 16x64 output tile.
// ---------------------------------------------------------------------------
__global__ __launch_bounds__(32) void outproj_kernel(
    const _Float16* __restrict__ z16, const _Float16* __restrict__ WoT,
    const float* __restrict__ bo, float* __restrict__ out) {
  __shared__ float ostg[16 * 64];
  int lane = threadIdx.x;
  int b = blockIdx.x;
  int ctb = b & 7;
  int rt  = b >> 3;
  int col = lane & 15, rhi = (lane >> 4) << 3;

  const _Float16* zb = z16 + (size_t)rt * 16 * EMB_;
  v8f acc[4];
#pragma unroll
  for (int j = 0; j < 4; ++j) {
    float bias = bo[ctb * 64 + 16 * j + col];
#pragma unroll
    for (int r = 0; r < 8; ++r) acc[j][r] = bias;
  }

  for (int kk = 0; kk < EMB_ / 32; ++kk) {
    v16h a = a_frag_f16(zb + 32 * kk, EMB_, lane);
#pragma unroll
    for (int j = 0; j < 4; ++j) {
      v16h bm = b_frag_f16_vec(WoT + (size_t)(32 * kk) * EMB_ + ctb * 64 + 16 * j,
                               EMB_, lane);
      acc[j] = wmma_f16(a, bm, acc[j]);
    }
  }

#pragma unroll
  for (int j = 0; j < 4; ++j)
#pragma unroll
    for (int r = 0; r < 8; ++r)
      ostg[(r + rhi) * 64 + 16 * j + col] = acc[j][r];
  {
    const int4* o4 = (const int4*)ostg;
    int row = lane >> 1, co = (lane & 1) * 32;
    float* op = out + ((size_t)rt * 16 + row) * EMB_ + ctb * 64 + co;
#pragma unroll
    for (int t = 0; t < 8; ++t)
      *(int4*)(op + 4 * t) = o4[lane * 8 + t];
  }
}

// ---------------------------------------------------------------------------
extern "C" void kernel_launch(void* const* d_in, const int* in_sizes, int n_in,
                              void* d_out, int out_size, void* d_ws, size_t ws_size,
                              hipStream_t stream) {
  const float* x  = (const float*)d_in[0];
  const float* Wq = (const float*)d_in[1];
  const float* bq = (const float*)d_in[2];
  const float* Wk = (const float*)d_in[3];
  const float* bk = (const float*)d_in[4];
  const float* Wv = (const float*)d_in[5];
  const float* bv = (const float*)d_in[6];
  const float* E  = (const float*)d_in[7];
  const float* Wo = (const float*)d_in[8];
  const float* bo = (const float*)d_in[9];
  float* out = (float*)d_out;

  const size_t nelem = (size_t)N_ * H_ * S_ * D_;
  _Float16* q16   = (_Float16*)d_ws;
  _Float16* kT    = q16 + nelem;
  _Float16* v16p  = kT + nelem;
  _Float16* z16   = v16p + nelem;
  _Float16* EtPad = z16 + (size_t)N_ * S_ * EMB_;
  _Float16* WoT   = EtPad + (size_t)D_ * SP_;

  prep_E_kernel<<<(D_ * SP_ + 255) / 256, 256, 0, stream>>>(E, EtPad);
  prep_Wo_kernel<<<(EMB_ * EMB_ + 255) / 256, 256, 0, stream>>>(Wo, WoT);
  proj_kernel<<<N_ * H_ * (S_ / 16), 32, 0, stream>>>(
      x, Wq, bq, Wk, bk, Wv, bv, q16, kT, v16p);
  attn_kernel<<<N_ * H_ * (S_ / 64), 128, 0, stream>>>(q16, kT, v16p, EtPad, z16);
  outproj_kernel<<<(N_ * S_ / 16) * (EMB_ / 64), 32, 0, stream>>>(z16, WoT, bo, out);
}
